// MultiHeadAttention_34583076667442
// MI455X (gfx1250) — matrix operand order repaired and checked by reading, not compiled, not run on hardware
//
#include <hip/hip_runtime.h>
#include <hip/hip_bf16.h>
#include <math.h>
#include <stdint.h>

typedef __attribute__((ext_vector_type(16))) __bf16 v16bf;
typedef __attribute__((ext_vector_type(8)))  __bf16 v8bf;
typedef __attribute__((ext_vector_type(8)))  float  v8f;

#define EMBED  1024
#define HEADS  16
#define HDIM   64
#define NBATCH 2
#define SEQ    2048
#define MROWS  (NBATCH*SEQ)      /* 4096 */
#define SCALEQ 0.125f            /* 64^-0.5 */
#define KCH    128               /* K chunk staged in LDS per barrier */
#define BPAD   16                /* 32B pad -> 72-DW row stride, aligned */

/* ---------------- f32 -> bf16 conversion (8 elems/thread) ---------------- */
__global__ void cvt8_f32_bf16(const float* __restrict__ src,
                              __bf16* __restrict__ dst, int n8) {
  int i = blockIdx.x * blockDim.x + threadIdx.x;
  if (i >= n8) return;
  const float* s = src + (size_t)i * 8;
  v8bf o;
#pragma unroll
  for (int j = 0; j < 8; ++j) o[j] = (__bf16)s[j];
  *(v8bf*)(dst + (size_t)i * 8) = o;
}

/* ---------------- Y = X @ W^T + b via WMMA bf16 ----------------
 * X: (M,K) bf16 row-major, W: (N,K) bf16 row-major (nn.Linear weight).
 * Block: 8 waves -> 128M x 64N; the 64xKCH W tile is shared by all 8
 * waves, staged once in LDS per K-chunk. Wave computes 16M x 64N.
 * mode 0: store bf16 into (Nb, H, L, D) head-split layout (q/k/v)
 * mode 1: store f32 row-major (final projection into d_out)          */
__global__ __launch_bounds__(256)
void lin_wmma(const __bf16* __restrict__ X, const __bf16* __restrict__ Wt,
              const float* __restrict__ bias, float* __restrict__ outF,
              __bf16* __restrict__ outHead, int M, int N, int K, int mode) {
  __shared__ __bf16 Bs[64][KCH + BPAD];     /* 18 KB */

  const int lane = threadIdx.x & 31;
  const int wave = threadIdx.x >> 5;
  const int hi   = lane >> 4;        /* lane-half: selects K sub-block */
  const int ln   = lane & 15;
  const int n0   = blockIdx.x * 64;
  const int m0   = blockIdx.y * 128 + wave * 16;

  v8f acc[4] = {};
  const __bf16* pArow = X + (size_t)(m0 + ln) * K;

  for (int kc = 0; kc < K; kc += KCH) {
    __syncthreads();                 /* protect previous chunk's readers */
    /* cooperative stage of W[n0..n0+63][kc..kc+127]: 1024 x 16B chunks */
#pragma unroll
    for (int j = 0; j < 4; ++j) {
      const int c    = threadIdx.x + j * 256;   /* 0..1023 */
      const int nrow = c >> 4;                  /* 16 chunks per row   */
      const int kcol = (c & 15) * 8;
      *(v8bf*)&Bs[nrow][kcol] =
          *(const v8bf*)(Wt + (size_t)(n0 + nrow) * K + kc + kcol);
    }
    __syncthreads();
    if (kc + KCH < K)                 /* pull next A chunk toward L2/L0 */
      __builtin_prefetch(pArow + kc + KCH, 0, 0);

#pragma unroll
    for (int k0 = 0; k0 < KCH; k0 += 32) {
      /* A 16x32: lane-half 0 holds K {0..15}, half 1 {16..31} */
      v16bf a;
      {
        const __bf16* p = pArow + kc + k0 + hi * 16;
        v8bf lo = *(const v8bf*)p;
        v8bf up = *(const v8bf*)(p + 8);
#pragma unroll
        for (int e = 0; e < 8; ++e) { a[e] = lo[e]; a[e + 8] = up[e]; }
      }
#pragma unroll
      for (int nt = 0; nt < 4; ++nt) {
        v16bf b = *(const v16bf*)&Bs[nt * 16 + ln][k0 + hi * 16];
        acc[nt] = __builtin_amdgcn_wmma_f32_16x16x32_bf16(
            false, a, false, b, (short)0, acc[nt], false, false);
      }
    }
  }

#pragma unroll
  for (int nt = 0; nt < 4; ++nt) {
    const int col = n0 + nt * 16 + ln;
    const float bv = bias[col];
#pragma unroll
    for (int r = 0; r < 8; ++r) {
      const int row = m0 + r + 8 * hi;   /* C layout: VGPR r -> M=r (+8) */
      const float val = acc[nt][r] + bv;
      if (mode == 0) {
        const int nb = row >> 11, l = row & (SEQ - 1);
        const int h = col >> 6, d = col & (HDIM - 1);
        outHead[(((size_t)(nb * HEADS + h)) * SEQ + l) * HDIM + d] = (__bf16)val;
      } else {
        outF[(size_t)row * N + col] = val;
      }
    }
  }
}

/* async 16B global -> LDS copy (gfx1250 ASYNCcnt path) */
__device__ __forceinline__ void async_g2l_b128(const void* gsrc, void* ldst) {
  uint32_t loff = (uint32_t)(uintptr_t)ldst;   /* low 32b = LDS byte offset */
  asm volatile("global_load_async_to_lds_b128 %0, %1, off"
               :: "v"(loff), "v"(gsrc) : "memory");
}
__device__ __forceinline__ void wait_async0() {
  asm volatile("s_wait_asynccnt 0x0" ::: "memory");
}

/* ---------------- flash-style attention, two-pass ----------------
 * grid.x = Nb*H (32), grid.y = L/128 (16). 8 waves: wave owns 16 query rows.
 * Pass 1: online row max/sum over S. Pass 2: recompute scores, write
 * normalized w to d_out once, accumulate O += w @ V via WMMA.          */
__global__ __launch_bounds__(256)
void attn_wmma(const __bf16* __restrict__ q, const __bf16* __restrict__ k,
               const __bf16* __restrict__ v, float* __restrict__ wOut,
               __bf16* __restrict__ attnOut) {
  __shared__ __bf16 Ks[32][64];        /* K tile, row-major (s,d)       4 KB */
  __shared__ __bf16 Vt[64][32];        /* V tile transposed (d,s)       4 KB */
  __shared__ __bf16 Wl[8][16][32];     /* per-wave w tile re-layout     8 KB */

  const int lane = threadIdx.x & 31;
  const int wave = threadIdx.x >> 5;
  const int hi   = lane >> 4;
  const int ln   = lane & 15;
  const int nh   = blockIdx.x;
  const int l0   = blockIdx.y * 128 + wave * 16;

  const __bf16* qb = q + (size_t)nh * SEQ * HDIM;
  const __bf16* kb = k + (size_t)nh * SEQ * HDIM;
  const __bf16* vb = v + (size_t)nh * SEQ * HDIM;

  /* Q rows for this wave as two A operands (d=0..31, d=32..63) */
  v16bf aQ[2];
  {
    const __bf16* pq = qb + (size_t)(l0 + ln) * HDIM;
#pragma unroll
    for (int c = 0; c < 2; ++c) {
      const __bf16* p = pq + c * 32 + hi * 16;
      v8bf lo = *(const v8bf*)p;
      v8bf up = *(const v8bf*)(p + 8);
#pragma unroll
      for (int e = 0; e < 8; ++e) { aQ[c][e] = lo[e]; aQ[c][e + 8] = up[e]; }
    }
  }

  float m[8], lsum[8];
#pragma unroll
  for (int r = 0; r < 8; ++r) { m[r] = -__builtin_inff(); lsum[r] = 0.f; }

  const int f  = threadIdx.x * 8;      /* cooperative staging: 256*8 = 32*64 */
  const int sr = f >> 6, dc = f & 63;

  /* ---------------- pass 1 ---------------- */
  for (int s0 = 0; s0 < SEQ; s0 += 32) {
    async_g2l_b128(kb + (size_t)(s0 + sr) * HDIM + dc, &Ks[sr][dc]);
    wait_async0();
    __syncthreads();
#pragma unroll
    for (int t = 0; t < 2; ++t) {
      v8f acc = {};
#pragma unroll
      for (int c = 0; c < 2; ++c) {
        v16bf b = *(const v16bf*)&Ks[t * 16 + ln][c * 32 + hi * 16];
        acc = __builtin_amdgcn_wmma_f32_16x16x32_bf16(
            false, aQ[c], false, b, (short)0, acc, false, false);
      }
#pragma unroll
      for (int r = 0; r < 8; ++r) {
        float s = acc[r] * SCALEQ;
        float tm = s;
#pragma unroll
        for (int msk = 1; msk < 16; msk <<= 1) tm = fmaxf(tm, __shfl_xor(tm, msk, 32));
        float mn = fmaxf(m[r], tm);
        float ew = __expf(s - mn);
#pragma unroll
        for (int msk = 1; msk < 16; msk <<= 1) ew += __shfl_xor(ew, msk, 32);
        lsum[r] = lsum[r] * __expf(m[r] - mn) + ew;
        m[r] = mn;
      }
    }
    __syncthreads();
  }
  float linv[8];
#pragma unroll
  for (int r = 0; r < 8; ++r) linv[r] = 1.f / lsum[r];  /* lsum >= 1, guard moot */

  /* ---------------- pass 2 ---------------- */
  v8f accO[4] = {};
  float* wRow = wOut + (size_t)nh * SEQ * SEQ;
  for (int s0 = 0; s0 < SEQ; s0 += 32) {
    async_g2l_b128(kb + (size_t)(s0 + sr) * HDIM + dc, &Ks[sr][dc]);
    {
      v8bf vv = *(const v8bf*)(vb + (size_t)(s0 + sr) * HDIM + dc);
#pragma unroll
      for (int j = 0; j < 8; ++j) Vt[dc + j][sr] = vv[j];
    }
    wait_async0();
    __syncthreads();
#pragma unroll
    for (int t = 0; t < 2; ++t) {
      v8f acc = {};
#pragma unroll
      for (int c = 0; c < 2; ++c) {
        v16bf b = *(const v16bf*)&Ks[t * 16 + ln][c * 32 + hi * 16];
        acc = __builtin_amdgcn_wmma_f32_16x16x32_bf16(
            false, aQ[c], false, b, (short)0, acc, false, false);
      }
#pragma unroll
      for (int r = 0; r < 8; ++r) {
        float s = acc[r] * SCALEQ;
        float w = __expf(s - m[r]) * linv[r];
        const int row = l0 + r + 8 * hi;
        wRow[(size_t)row * SEQ + (s0 + t * 16 + ln)] = w;     /* write w once */
        Wl[wave][r + 8 * hi][t * 16 + ln] = (__bf16)w;
      }
    }
    /* re-gather w tile as A operand (wave-private LDS region, in-order DS) */
    v16bf aW;
    {
      const __bf16* p = &Wl[wave][ln][hi * 16];
      v8bf lo = *(const v8bf*)p;
      v8bf up = *(const v8bf*)(p + 8);
#pragma unroll
      for (int e = 0; e < 8; ++e) { aW[e] = lo[e]; aW[e + 8] = up[e]; }
    }
#pragma unroll
    for (int dt = 0; dt < 4; ++dt) {
      v16bf bV = *(const v16bf*)&Vt[dt * 16 + ln][hi * 16];
      accO[dt] = __builtin_amdgcn_wmma_f32_16x16x32_bf16(
          false, aW, false, bV, (short)0, accO[dt], false, false);
    }
    __syncthreads();
  }

  /* epilogue: attn (Nb, L, E) bf16, col = h*64 + d */
  const int nb = nh >> 4, h = nh & (HEADS - 1);
#pragma unroll
  for (int dt = 0; dt < 4; ++dt)
#pragma unroll
    for (int r = 0; r < 8; ++r) {
      const int row = l0 + r + 8 * hi;
      const int col = h * HDIM + dt * 16 + ln;
      attnOut[(size_t)(nb * SEQ + row) * EMBED + col] = (__bf16)accO[dt][r];
    }
}

/* ---------------- host-side orchestration ---------------- */
extern "C" void kernel_launch(void* const* d_in, const int* in_sizes, int n_in,
                              void* d_out, int out_size, void* d_ws, size_t ws_size,
                              hipStream_t stream) {
  const float* query = (const float*)d_in[0];
  const float* key_  = (const float*)d_in[1];
  const float* value = (const float*)d_in[2];
  const float* Wq = (const float*)d_in[3]; const float* bq = (const float*)d_in[4];
  const float* Wk = (const float*)d_in[5]; const float* bk = (const float*)d_in[6];
  const float* Wv = (const float*)d_in[7]; const float* bv = (const float*)d_in[8];
  const float* Wo = (const float*)d_in[9]; const float* bo = (const float*)d_in[10];

  const size_t XE = (size_t)MROWS * EMBED;   /* 4,194,304 */
  const size_t WE = (size_t)EMBED * EMBED;   /* 1,048,576 */
  if (ws_size < (3 * XE + 4 * WE + 3 * XE + XE) * sizeof(__bf16)) return;

  __bf16* ws = (__bf16*)d_ws;
  __bf16* Xq = ws;            __bf16* Xk = Xq + XE;  __bf16* Xv = Xk + XE;
  __bf16* Wqb = Xv + XE;      __bf16* Wkb = Wqb + WE;
  __bf16* Wvb = Wkb + WE;     __bf16* Wob = Wvb + WE;
  __bf16* qh  = Wob + WE;     __bf16* kh = qh + XE;  __bf16* vh = kh + XE;
  __bf16* attnB = vh + XE;

  float* outF = (float*)d_out;
  float* wOut = outF + XE;                   /* (out, w) concatenated flat */

  const int TB = 256;
  cvt8_f32_bf16<<<(int)(XE / 8 + TB - 1) / TB, TB, 0, stream>>>(query, Xq, (int)(XE / 8));
  cvt8_f32_bf16<<<(int)(XE / 8 + TB - 1) / TB, TB, 0, stream>>>(key_,  Xk, (int)(XE / 8));
  cvt8_f32_bf16<<<(int)(XE / 8 + TB - 1) / TB, TB, 0, stream>>>(value, Xv, (int)(XE / 8));
  cvt8_f32_bf16<<<(int)(WE / 8 + TB - 1) / TB, TB, 0, stream>>>(Wq, Wqb, (int)(WE / 8));
  cvt8_f32_bf16<<<(int)(WE / 8 + TB - 1) / TB, TB, 0, stream>>>(Wk, Wkb, (int)(WE / 8));
  cvt8_f32_bf16<<<(int)(WE / 8 + TB - 1) / TB, TB, 0, stream>>>(Wv, Wvb, (int)(WE / 8));
  cvt8_f32_bf16<<<(int)(WE / 8 + TB - 1) / TB, TB, 0, stream>>>(Wo, Wob, (int)(WE / 8));

  dim3 lg(EMBED / 64, MROWS / 128);          /* (16, 32) */
  lin_wmma<<<lg, TB, 0, stream>>>(Xq, Wqb, bq, nullptr, qh, MROWS, EMBED, EMBED, 0);
  lin_wmma<<<lg, TB, 0, stream>>>(Xk, Wkb, bk, nullptr, kh, MROWS, EMBED, EMBED, 0);
  lin_wmma<<<lg, TB, 0, stream>>>(Xv, Wvb, bv, nullptr, vh, MROWS, EMBED, EMBED, 0);

  attn_wmma<<<dim3(NBATCH * HEADS, SEQ / 128), TB, 0, stream>>>(qh, kh, vh, wOut, attnB);

  lin_wmma<<<lg, TB, 0, stream>>>(attnB, Wob, bo, outF, nullptr, MROWS, EMBED, EMBED, 1);
}
